// Semantic_70342974374320
// MI455X (gfx1250) — matrix operand order repaired and checked by reading, not compiled, not run on hardware
//
#include <hip/hip_runtime.h>

typedef __attribute__((ext_vector_type(16))) _Float16 v16h;
typedef __attribute__((ext_vector_type(8)))  _Float16 v8h;
typedef __attribute__((ext_vector_type(8)))  float    v8f;

#define CONV_WAVES_PER_BLOCK 8
#define CONV_TILES_PER_WAVE  4

// ---------------------------------------------------------------------------
// utility: grid-stride zero
// ---------------------------------------------------------------------------
__global__ void zero_f32(float* __restrict__ p, long n) {
    long i = (long)blockIdx.x * blockDim.x + threadIdx.x;
    long stride = (long)gridDim.x * blockDim.x;
    for (; i < n; i += stride) p[i] = 0.0f;
}

// ---------------------------------------------------------------------------
// convert W [K][ci][co] f32 -> Wt [K][co][ci] f16  (transpose so WMMA B-tile
// reads are contiguous 128-bit loads)
// ---------------------------------------------------------------------------
__global__ void wconv_kernel(const float* __restrict__ W1, const float* __restrict__ W2,
                             _Float16* __restrict__ Wt1, _Float16* __restrict__ Wt2,
                             int total) {
    int i = blockIdx.x * blockDim.x + threadIdx.x;
    if (i >= total) return;
    int k   = i >> 10;           // / 1024
    int rem = i & 1023;
    int co  = rem >> 5;
    int ci  = rem & 31;
    long src = (long)k * 1024 + (long)ci * 32 + co;
    Wt1[i] = (_Float16)W1[src];
    Wt2[i] = (_Float16)W2[src];
}

// ---------------------------------------------------------------------------
// BatchNorm stats: per-channel sum / sumsq over N rows (C = 32 channels).
// Stride is a multiple of 32 -> each thread's channel is fixed.
// LDS f32 atomics per block, one global atomic per channel per block.
// ---------------------------------------------------------------------------
__global__ void bn_stats(const float* __restrict__ x, float* __restrict__ stats, long total) {
    __shared__ float s_sum[32];
    __shared__ float s_sq[32];
    int t = threadIdx.x;
    if (t < 32) { s_sum[t] = 0.0f; s_sq[t] = 0.0f; }
    __syncthreads();
    long i = (long)blockIdx.x * blockDim.x + t;
    long stride = (long)gridDim.x * blockDim.x;
    int c = (int)(i & 31);
    float ls = 0.0f, lq = 0.0f;
    for (; i < total; i += stride) {
        float v = x[i];
        ls += v;
        lq += v * v;
    }
    atomicAdd(&s_sum[c], ls);
    atomicAdd(&s_sq[c], lq);
    __syncthreads();
    if (t < 32) {
        atomicAdd(&stats[t], s_sum[t]);
        atomicAdd(&stats[32 + t], s_sq[t]);
    }
}

// ---------------------------------------------------------------------------
// y = f16( relu( (x - mu) * rsqrt(var + eps) ) )
// ---------------------------------------------------------------------------
__global__ void bn_apply(const float* __restrict__ x, const float* __restrict__ stats,
                         _Float16* __restrict__ y, long total, float invN) {
    long i = (long)blockIdx.x * blockDim.x + threadIdx.x;
    long stride = (long)gridDim.x * blockDim.x;
    for (; i < total; i += stride) {
        int c = (int)(i & 31);
        float mean = stats[c] * invN;
        float var  = stats[32 + c] * invN - mean * mean;
        float rs   = rsqrtf(var + 1e-4f);
        float v    = (x[i] - mean) * rs;
        y[i] = (_Float16)fmaxf(v, 0.0f);
    }
}

// ---------------------------------------------------------------------------
// Submanifold conv: gather 16 pairs -> [16x32] f16 A, W[k] as two 32x16 B
// tiles (hoisted per-k), two v_wmma_f32_16x16x32_f16 per tile, scatter-add D
// with global f32 atomics. One wave = CONV_TILES_PER_WAVE tiles; k = blockIdx.y
// (no integer division); pair indices broadcast via wave32 shuffles (no LDS
// round trip / no block barrier in the hot loop).
// ---------------------------------------------------------------------------
__global__ __launch_bounds__(CONV_WAVES_PER_BLOCK * 32)
void subm_conv_wmma(const _Float16* __restrict__ y, const _Float16* __restrict__ Wt,
                    const int* __restrict__ pair_in, const int* __restrict__ pair_out,
                    float* __restrict__ out, int P, int tiles_per_k) {
    const int wave = threadIdx.x >> 5;
    const int lane = threadIdx.x & 31;
    const int m    = lane & 15;          // row / column index within 16-tile
    const int hi   = lane >> 4;          // half-wave select
    const int k    = blockIdx.y;
    const long kbase = (long)k * P;

    // ---- B (32x16 f16, K x N), hoisted: lane holds column N = m; lanes 0-15
    // carry K=0..15, lanes 16-31 carry K=16..31. Wt is [k][co][ci].
    const _Float16* wb  = Wt + (long)k * 1024;
    const _Float16* cb0 = wb + (long)m * 32 + hi * 16;          // co = m
    const _Float16* cb1 = wb + (long)(m + 16) * 32 + hi * 16;   // co = m + 16
    v16h b0, b1;
    {
        v8h t0 = *(const v8h*)(cb0);
        v8h t1 = *(const v8h*)(cb0 + 8);
        v8h u0 = *(const v8h*)(cb1);
        v8h u1 = *(const v8h*)(cb1 + 8);
#pragma unroll
        for (int i = 0; i < 8; ++i) {
            b0[i] = t0[i]; b0[8 + i] = t1[i];
            b1[i] = u0[i]; b1[8 + i] = u1[i];
        }
    }

    int tile = (blockIdx.x * CONV_WAVES_PER_BLOCK + wave) * CONV_TILES_PER_WAVE;
    int tile_end = tile + CONV_TILES_PER_WAVE;
    if (tile_end > tiles_per_k) tile_end = tiles_per_k;

    for (; tile < tile_end; ++tile) {
        const int p = tile * 16 + m;
        int pin  = (p < P) ? pair_in[kbase + p] : -1;
        int pout = (p < P && pin >= 0) ? pair_out[kbase + p] : 0;

        // prefetch next tile's pair indices (global_prefetch_b8)
        if (p + 16 < P) {
            __builtin_prefetch(&pair_in[kbase + p + 16], 0, 3);
            __builtin_prefetch(&pair_out[kbase + p + 16], 0, 3);
        }

        // ---- A (16x32 f16): lane L holds row M = L%16.
        // lanes 0-15: K = 0..15 ; lanes 16-31: K = 16..31
        v16h a = {};
        if (pin >= 0) {
            const _Float16* row = y + (long)pin * 32;
            v8h x0 = *(const v8h*)(row + hi * 16);
            v8h x1 = *(const v8h*)(row + hi * 16 + 8);
#pragma unroll
            for (int i = 0; i < 8; ++i) { a[i] = x0[i]; a[8 + i] = x1[i]; }
        }

        // ---- D = A x B (+0): two WMMAs cover output channels 0..15 / 16..31
        v8f acc0 = {}, acc1 = {};
        acc0 = __builtin_amdgcn_wmma_f32_16x16x32_f16(false, a, false, b0,
                                                      (short)0, acc0, false, false);
        acc1 = __builtin_amdgcn_wmma_f32_16x16x32_f16(false, a, false, b1,
                                                      (short)0, acc1, false, false);

        // ---- scatter: D element v -> row M = v + hi*8, channel m / m+16.
        // Row M's indices live in lane M; broadcast with wave32 shuffles.
#pragma unroll
        for (int v = 0; v < 8; ++v) {
            const int M = v + hi * 8;
            int rin  = __shfl(pin,  M, 32);
            int rout = __shfl(pout, M, 32);
            if (rin >= 0) {
                long o = (long)rout * 32;
                atomicAdd(out + o + m,      acc0[v]);
                atomicAdd(out + o + 16 + m, acc1[v]);
            }
        }
    }
}

// ---------------------------------------------------------------------------
// out = h + feats  (residual)
// ---------------------------------------------------------------------------
__global__ void residual_add(const float* __restrict__ h, const float* __restrict__ f,
                             float* __restrict__ out, long total) {
    long i = (long)blockIdx.x * blockDim.x + threadIdx.x;
    long stride = (long)gridDim.x * blockDim.x;
    for (; i < total; i += stride) out[i] = h[i] + f[i];
}

// ---------------------------------------------------------------------------
// downsample scatter-mean accumulate + finalize
// ---------------------------------------------------------------------------
__global__ void ds_scatter(const float* __restrict__ xyz, const int* __restrict__ batch,
                           const int* __restrict__ pin, const int* __restrict__ pout,
                           float* __restrict__ sums, float* __restrict__ bsum,
                           float* __restrict__ cnt, int P2) {
    int i = blockIdx.x * blockDim.x + threadIdx.x;
    if (i >= P2) return;
    int in = pin[i];
    if (in < 0) return;
    int o = pout[i];
    atomicAdd(&sums[(long)o * 3 + 0], xyz[(long)in * 3 + 0]);
    atomicAdd(&sums[(long)o * 3 + 1], xyz[(long)in * 3 + 1]);
    atomicAdd(&sums[(long)o * 3 + 2], xyz[(long)in * 3 + 2]);
    atomicAdd(&bsum[o], (float)batch[in]);
    atomicAdd(&cnt[o], 1.0f);
}

__global__ void ds_finalize(const float* __restrict__ sums, const float* __restrict__ bsum,
                            const float* __restrict__ cnt, float* __restrict__ oxyz,
                            float* __restrict__ obatch, int M) {
    int i = blockIdx.x * blockDim.x + threadIdx.x;
    if (i >= M) return;
    float inv = 1.0f / fmaxf(cnt[i], 1.0f);
    oxyz[(long)i * 3 + 0] = sums[(long)i * 3 + 0] * inv;
    oxyz[(long)i * 3 + 1] = sums[(long)i * 3 + 1] * inv;
    oxyz[(long)i * 3 + 2] = sums[(long)i * 3 + 2] * inv;
    obatch[i] = bsum[i] * inv;
}

// ---------------------------------------------------------------------------
extern "C" void kernel_launch(void* const* d_in, const int* in_sizes, int n_in,
                              void* d_out, int out_size, void* d_ws, size_t ws_size,
                              hipStream_t stream) {
    const float* feats    = (const float*)d_in[0];
    const float* xyz      = (const float*)d_in[1];
    const int*   batch    = (const int*)d_in[2];
    const int*   pair_in  = (const int*)d_in[3];
    const int*   pair_out = (const int*)d_in[4];
    const int*   ds_pin   = (const int*)d_in[5];
    const int*   ds_pout  = (const int*)d_in[6];
    const float* W1       = (const float*)d_in[7];
    const float* W2       = (const float*)d_in[8];

    const int K  = 27;
    const int N  = in_sizes[0] / 32;
    const int P  = in_sizes[3] / K;
    const int P2 = in_sizes[5];
    const int M  = N / 8;
    const long NC = (long)N * 32;

    // workspace layout
    char* ws = (char*)d_ws;
    size_t off = 0;
    auto salloc = [&](size_t bytes) -> void* {
        void* p = ws + off;
        off = (off + bytes + 255) & ~(size_t)255;
        return p;
    };
    float*    h     = (float*)salloc((size_t)NC * 4);
    _Float16* y     = (_Float16*)salloc((size_t)NC * 2);
    _Float16* Wt1   = (_Float16*)salloc((size_t)K * 1024 * 2);
    _Float16* Wt2   = (_Float16*)salloc((size_t)K * 1024 * 2);
    float*    stats = (float*)salloc(64 * 4);
    float*    sums  = (float*)salloc((size_t)M * 3 * 4);
    float*    bsum  = (float*)salloc((size_t)M * 4);
    float*    cnt   = (float*)salloc((size_t)M * 4);

    float* out_feats = (float*)d_out;
    float* out_xyz   = out_feats + NC;
    float* out_batch = out_xyz + (long)M * 3;

    const int tiles_per_k   = (P + 15) / 16;
    const int tiles_per_blk = CONV_WAVES_PER_BLOCK * CONV_TILES_PER_WAVE;
    const dim3 conv_grid((tiles_per_k + tiles_per_blk - 1) / tiles_per_blk, K);
    const float invN = 1.0f / (float)N;
    const int apply_blocks = (int)((NC + 255) / 256);

    // weights -> f16 transposed (once per launch; tiny)
    int wtot = K * 1024;
    wconv_kernel<<<(wtot + 255) / 256, 256, 0, stream>>>(W1, W2, Wt1, Wt2, wtot);

    // --- BN1 -> y1 (f16)
    zero_f32<<<1, 64, 0, stream>>>(stats, 64);
    bn_stats<<<512, 256, 0, stream>>>(feats, stats, NC);
    bn_apply<<<apply_blocks, 256, 0, stream>>>(feats, stats, y, NC, invN);

    // --- conv1: h = subm(y1, W1)
    zero_f32<<<2048, 256, 0, stream>>>(h, NC);
    subm_conv_wmma<<<conv_grid, CONV_WAVES_PER_BLOCK * 32, 0, stream>>>(
        y, Wt1, pair_in, pair_out, h, P, tiles_per_k);

    // --- BN2 -> y2 (f16)
    zero_f32<<<1, 64, 0, stream>>>(stats, 64);
    bn_stats<<<512, 256, 0, stream>>>(h, stats, NC);
    bn_apply<<<apply_blocks, 256, 0, stream>>>(h, stats, y, NC, invN);

    // --- conv2: h = subm(y2, W2)   (h reused, re-zeroed)
    zero_f32<<<2048, 256, 0, stream>>>(h, NC);
    subm_conv_wmma<<<conv_grid, CONV_WAVES_PER_BLOCK * 32, 0, stream>>>(
        y, Wt2, pair_in, pair_out, h, P, tiles_per_k);

    // --- residual
    residual_add<<<2048, 256, 0, stream>>>(h, feats, out_feats, NC);

    // --- downsample scatter-mean
    zero_f32<<<256, 256, 0, stream>>>(sums, (long)M * 3);
    zero_f32<<<64, 256, 0, stream>>>(bsum, M);
    zero_f32<<<64, 256, 0, stream>>>(cnt, M);
    ds_scatter<<<(P2 + 255) / 256, 256, 0, stream>>>(xyz, batch, ds_pin, ds_pout,
                                                     sums, bsum, cnt, P2);
    ds_finalize<<<(M + 255) / 256, 256, 0, stream>>>(sums, bsum, cnt,
                                                     out_xyz, out_batch, M);
}
